// cap_bdg_67980742361948
// MI455X (gfx1250) — compile-verified, matrix operand order checked by reading, not run on hardware
//
#include <hip/hip_runtime.h>
#include <math.h>

// ---------------- problem constants ----------------
#define Bb 8
#define Cc 64
#define Nn 16
#define HH 96
#define WWID 96
#define HWp 9216          // 96*96
#define BN 128            // Bb*Nn
#define EPSV 1e-5f

// ---------------- d_out layout (floats) ----------------
static const size_t OUT_OPOUT = 0;                       // 8*16*64*9216
static const size_t OUT_MSK   = 75497472;                // 8*16*9216
static const size_t OUT_REC   = 76677120;                // 8*3*9216
static const size_t OUT_LOSS  = 76898304;                // 1
static const size_t OUT_COF   = 76898305;                // 8*16*9216

// ---------------- ws layout (float offsets) ----------------
static const size_t WS_TGSUM = 0;          // 512
static const size_t WS_TGSS  = 512;        // 512
static const size_t WS_SRSUM = 1024;       // 512
static const size_t WS_OP    = 1536;       // 128*64
static const size_t WS_SC1   = 9728;       // 128*64
static const size_t WS_SH1   = 17920;      // 128*64
static const size_t WS_GN2ST = 26112;      // 128*16*2
static const size_t WS_SC2   = 30208;      // 128*64
static const size_t WS_SH2   = 38400;      // 128*64
static const size_t WS_IN1SC = 46592;      // 128
static const size_t WS_IN1SH = 46720;      // 128
static const size_t WS_IN2ST = 46848;      // 8*16*2
static const size_t WS_IN2SC = 47104;      // 128
static const size_t WS_IN2SH = 47232;      // 128
static const size_t WS_G2    = 47360;      // 8*16*9216
static const size_t WS_WA    = 1227008;    // 36864 ushorts (18432 floats)
static const size_t WS_Y1    = 1245440;    // 75497472 ushorts (bf16 y1)

typedef __attribute__((ext_vector_type(16))) __bf16 v16bf;
typedef __attribute__((ext_vector_type(8)))  float  v8f;

union Frag16 { v16bf v; uint4 q[2]; unsigned u[8]; unsigned short h[16]; };

__device__ __forceinline__ unsigned short f2bf(float f) {
    unsigned u = __float_as_uint(f);
    u += 0x7FFFu + ((u >> 16) & 1u);
    return (unsigned short)(u >> 16);
}
__device__ __forceinline__ float bf2f(unsigned short s) {
    return __uint_as_float(((unsigned)s) << 16);
}

// =========== zero accumulators ===========
__global__ void k_zero(float* gn2st, float* in2st, float* loss) {
    int i = blockIdx.x * 256 + threadIdx.x;
    if (i < 4096) gn2st[i] = 0.f;
    if (i < 256)  in2st[i] = 0.f;
    if (i == 0)   *loss = 0.f;
}

// =========== per-(b,c) sums of tg and sr ===========
__global__ void k_stats(const float* __restrict__ tg, const float* __restrict__ sr,
                        float* tg_sum, float* tg_ss, float* sr_sum) {
    int bc = blockIdx.x;                 // 0..511
    int tid = threadIdx.x;
    const float* trow = tg + (size_t)bc * HWp;
    const float* srow = sr + (size_t)bc * HWp;
    float s = 0.f, ss = 0.f, s2 = 0.f;
    for (int p = tid; p < HWp; p += 256) {
        float v = trow[p]; s += v; ss += v * v; s2 += srow[p];
    }
    __shared__ float r0[256], r1[256], r2[256];
    r0[tid] = s; r1[tid] = ss; r2[tid] = s2; __syncthreads();
    for (int st = 128; st > 0; st >>= 1) {
        if (tid < st) { r0[tid] += r0[tid+st]; r1[tid] += r1[tid+st]; r2[tid] += r2[tid+st]; }
        __syncthreads();
    }
    if (tid == 0) { tg_sum[bc] = r0[0]; tg_ss[bc] = r1[0]; sr_sum[bc] = r2[0]; }
}

// =========== dynamic routing (s never materialized) ===========
__global__ void k_routing(const float* __restrict__ sr, const float* __restrict__ w_w,
                          const float* __restrict__ w_b, const float* __restrict__ sr_sum,
                          float* op_ws, float* cof_out) {
    int bn = blockIdx.x, b = bn >> 4, n = bn & 15, tid = threadIdx.x;
    __shared__ float wn[64 * 64];
    __shared__ float cof[HWp];
    __shared__ float red[256];
    __shared__ float opv[64], opn[64], vv[64], rr[64];
    __shared__ float bd_s;
    for (int i = tid; i < 4096; i += 256)
        wn[i] = w_w[(size_t)(n * 64 + (i >> 6)) * 64 + (i & 63)];
    __syncthreads();
    if (tid < 64) {
        float a = 0.f;
        for (int c2 = 0; c2 < 64; c2++) a += wn[tid * 64 + c2] * sr_sum[b * 64 + c2];
        opv[tid] = a + (float)HWp * w_b[n * 64 + tid];
    }
    __syncthreads();
    const float* srb = sr + (size_t)b * 64 * HWp;
    for (int it = 0; it < 3; ++it) {
        float part = (tid < 64) ? opv[tid] * opv[tid] : 0.f;
        red[tid] = part; __syncthreads();
        for (int st = 128; st > 0; st >>= 1) { if (tid < st) red[tid] += red[tid+st]; __syncthreads(); }
        float nrm = fmaxf(sqrtf(red[0]), 1e-12f); __syncthreads();
        if (tid < 64) opn[tid] = opv[tid] / nrm;
        __syncthreads();
        if (tid < 64) {
            float a = 0.f;
            for (int c = 0; c < 64; c++) a += wn[c * 64 + tid] * opn[c];
            vv[tid] = a;
        }
        if (tid == 0) {
            float a = 0.f;
            for (int c = 0; c < 64; c++) a += w_b[n * 64 + c] * opn[c];
            bd_s = a;
        }
        __syncthreads();
        float bd = bd_s;
        float lmax = -1e30f;
        for (int p = tid; p < HWp; p += 256) {
            float d = bd;
            for (int c = 0; c < 64; c++) d += srb[(size_t)c * HWp + p] * vv[c];
            cof[p] = d; lmax = fmaxf(lmax, d);
        }
        red[tid] = lmax; __syncthreads();
        for (int st = 128; st > 0; st >>= 1) { if (tid < st) red[tid] = fmaxf(red[tid], red[tid+st]); __syncthreads(); }
        float mx = red[0]; __syncthreads();
        float lsum = 0.f;
        for (int p = tid; p < HWp; p += 256) { float e = __expf(cof[p] - mx); cof[p] = e; lsum += e; }
        red[tid] = lsum; __syncthreads();
        for (int st = 128; st > 0; st >>= 1) { if (tid < st) red[tid] += red[tid+st]; __syncthreads(); }
        float inv = 1.f / red[0]; __syncthreads();
        for (int p = tid; p < HWp; p += 256) {
            float cv = cof[p] * inv; cof[p] = cv;
            if (it == 2) cof_out[(size_t)bn * HWp + p] = cv;
        }
        __syncthreads();
        for (int c = 0; c < 64; c++) {
            float a = 0.f;
            for (int p = tid; p < HWp; p += 256) a += srb[(size_t)c * HWp + p] * cof[p];
            red[tid] = a; __syncthreads();
            for (int st = 128; st > 0; st >>= 1) { if (tid < st) red[tid] += red[tid+st]; __syncthreads(); }
            if (tid == 0) rr[c] = red[0];
            __syncthreads();
        }
        if (tid < 64) {
            float a = w_b[n * 64 + tid];
            for (int c2 = 0; c2 < 64; c2++) a += wn[tid * 64 + c2] * rr[c2];
            opv[tid] = a;
        }
        __syncthreads();
    }
    if (tid < 64) op_ws[bn * 64 + tid] = opv[tid];
}

// =========== analytic GN1 -> per-(bn,c) affine of tg ===========
__global__ void k_gn1prep(const float* tg_sum, const float* tg_ss, const float* op,
                          const float* g1, const float* b1, float* sc1, float* sh1) {
    int bn = blockIdx.x, b = bn >> 4, c = threadIdx.x;
    __shared__ float ms[64], es[64];
    float opc = op[bn * 64 + c];
    float ts = tg_sum[b * 64 + c], tss = tg_ss[b * 64 + c];
    ms[c] = ts + (float)HWp * opc;
    es[c] = tss + 2.f * opc * ts + (float)HWp * opc * opc;
    __syncthreads();
    int g4 = c & ~3;
    const float invM = 1.f / (4.f * (float)HWp);
    float mu  = (ms[g4] + ms[g4+1] + ms[g4+2] + ms[g4+3]) * invM;
    float ex2 = (es[g4] + es[g4+1] + es[g4+2] + es[g4+3]) * invM;
    float var = ex2 - mu * mu;
    float s = g1[c] * rsqrtf(var + EPSV);
    sc1[bn * 64 + c] = s;
    sh1[bn * 64 + c] = (opc - mu) * s + b1[c];
}

// =========== pack conv1 weights into WMMA A-fragments (bf16) ===========
__global__ void k_packw1(const float* __restrict__ c1w, unsigned short* wA) {
    int i = blockIdx.x * 256 + threadIdx.x;
    if (i >= 18 * 4 * 32 * 16) return;
    int e = i & 15, frag = i >> 4;
    int lane = frag & 31, og = (frag >> 5) & 3, kc = frag >> 7;
    int v = e >> 1, h = e & 1;
    int klocal = ((v >> 2) << 4) + ((lane >= 16) ? 8 : 0) + ((v & 3) << 1) + h;
    int ky = kc / 6, r6 = kc % 6, kx = r6 >> 1, cibase = (r6 & 1) * 32;
    int ci = cibase + klocal, o = og * 16 + (lane & 15);
    wA[i] = f2bf(c1w[((o * 64 + ci) * 3 + ky) * 3 + kx]);
}

// =========== conv1: implicit-GEMM 3x3 64->64 via v_wmma bf16 ===========
__global__ void __launch_bounds__(256) k_conv1(
        const float* __restrict__ tg, const float* __restrict__ sc1,
        const float* __restrict__ sh1, const float* __restrict__ c1b,
        const unsigned short* __restrict__ wA, unsigned short* __restrict__ y1,
        float* gn2st) {
    int blk = blockIdx.x;
    int y = blk % HH, bn = blk / HH, b = bn >> 4;
    int tid = threadIdx.x;
    __shared__ __align__(16) unsigned short in_s[3 * 98 * 64];   // [row][x][ci] bf16
    __shared__ float scs[64], shs[64], bias_s[64];
    if (tid < 64) { scs[tid] = sc1[bn * 64 + tid]; shs[tid] = sh1[bn * 64 + tid]; bias_s[tid] = c1b[tid]; }
    __syncthreads();
    // stage relu(GN1(tg + op)) row triple (with halo + zero pad) into LDS as bf16
    for (int i = tid; i < 3 * 98 * 64; i += 256) {
        int row = i / (64 * 98), rem = i % (64 * 98), ci = rem / 98, x = rem % 98;
        int gy = y + row - 1, gx = x - 1;
        unsigned short hv = 0;
        if ((unsigned)gy < 96u && (unsigned)gx < 96u) {
            float v = tg[((size_t)(b * 64 + ci)) * HWp + gy * 96 + gx];
            v = fmaxf(v * scs[ci] + shs[ci], 0.f);
            hv = f2bf(v);
        }
        in_s[(row * 98 + x) * 64 + ci] = hv;
    }
    __syncthreads();
    int lane = tid & 31, wv = tid >> 5;
    int og = wv & 3, pgbase = (wv >> 2) * 3;
    int hi8 = (lane >= 16) ? 8 : 0, lx = lane & 15;
    v8f acc[3];
    for (int t = 0; t < 3; t++) for (int r = 0; r < 8; r++) acc[t][r] = 0.f;
    for (int kc = 0; kc < 18; kc++) {
        int ky = kc / 6, r6 = kc % 6, kx = r6 >> 1, cibase = (r6 & 1) * 32;
        Frag16 a;
        const uint4* ap = (const uint4*)(wA + ((size_t)(kc * 4 + og) * 32 + lane) * 16);
        a.q[0] = ap[0]; a.q[1] = ap[1];
        for (int t = 0; t < 3; t++) {
            Frag16 bf;
            int xcol = (pgbase + t) * 16 + lx + kx;
            const unsigned short* base = &in_s[(ky * 98 + xcol) * 64 + cibase + hi8];
            bf.u[0] = *(const unsigned*)(base + 0);
            bf.u[1] = *(const unsigned*)(base + 2);
            bf.u[2] = *(const unsigned*)(base + 4);
            bf.u[3] = *(const unsigned*)(base + 6);
            bf.u[4] = *(const unsigned*)(base + 16);
            bf.u[5] = *(const unsigned*)(base + 18);
            bf.u[6] = *(const unsigned*)(base + 20);
            bf.u[7] = *(const unsigned*)(base + 22);
            acc[t] = __builtin_amdgcn_wmma_f32_16x16x32_bf16(
                false, a.v, false, bf.v, (short)0, acc[t], false, false);
        }
    }
    // epilogue: +bias, store y1 (bf16), fused GN2 partial stats
    float s0 = 0.f, ss0 = 0.f, s1 = 0.f, ss1 = 0.f;
    for (int t = 0; t < 3; t++) {
        int px = (pgbase + t) * 16 + lx;
        size_t colbase = (size_t)bn * 64 * HWp + (size_t)y * 96 + px;
        for (int r = 0; r < 8; r++) {
            int orow = og * 16 + hi8 + r;
            float val = acc[t][r] + bias_s[orow];
            y1[colbase + (size_t)orow * HWp] = f2bf(val);
            if (r < 4) { s0 += val; ss0 += val * val; }
            else       { s1 += val; ss1 += val * val; }
        }
    }
    for (int m = 1; m < 16; m <<= 1) {
        s0 += __shfl_xor(s0, m, 32); ss0 += __shfl_xor(ss0, m, 32);
        s1 += __shfl_xor(s1, m, 32); ss1 += __shfl_xor(ss1, m, 32);
    }
    if (lx == 0) {
        int g = og * 4 + ((lane >= 16) ? 2 : 0);
        atomicAdd(&gn2st[(bn * 16 + g) * 2 + 0], s0);
        atomicAdd(&gn2st[(bn * 16 + g) * 2 + 1], ss0);
        atomicAdd(&gn2st[(bn * 16 + g + 1) * 2 + 0], s1);
        atomicAdd(&gn2st[(bn * 16 + g + 1) * 2 + 1], ss1);
    }
}

// =========== GN2 finalize -> per-(bn,c) affine ===========
__global__ void k_gn2fin(const float* gn2st, const float* g2, const float* b2,
                         float* sc2, float* sh2) {
    int bn = blockIdx.x, c = threadIdx.x, g = c >> 2;
    float sum = gn2st[(bn * 16 + g) * 2], ss = gn2st[(bn * 16 + g) * 2 + 1];
    const float invM = 1.f / (4.f * (float)HWp);
    float mu = sum * invM, var = ss * invM - mu * mu;
    float s = g2[c] * rsqrtf(var + EPSV);
    sc2[bn * 64 + c] = s;
    sh2[bn * 64 + c] = b2[c] - mu * s;
}

// =========== conv2: 3x3 64->1 from bf16 y1 ===========
__global__ void __launch_bounds__(256) k_conv2(
        const unsigned short* __restrict__ y1, const float* sc2, const float* sh2,
        const float* __restrict__ c2w, const float* c2b, float* mskout) {
    int blk = blockIdx.x;
    int y = blk % HH, bn = blk / HH;
    int tid = threadIdx.x;
    __shared__ __align__(16) unsigned short x2[3 * 64 * 98];   // [row][ci][x]
    __shared__ float w2s[576], scs[64], shs[64];
    if (tid < 64) { scs[tid] = sc2[bn * 64 + tid]; shs[tid] = sh2[bn * 64 + tid]; }
    for (int i = tid; i < 576; i += 256) w2s[i] = c2w[i];
    __syncthreads();
    for (int i = tid; i < 3 * 64 * 98; i += 256) {
        int row = i / (64 * 98), rem = i % (64 * 98), ci = rem / 98, x = rem % 98;
        int gy = y + row - 1, gx = x - 1;
        unsigned short hv = 0;
        if ((unsigned)gy < 96u && (unsigned)gx < 96u) {
            float v = bf2f(y1[((size_t)(bn * 64 + ci)) * HWp + gy * 96 + gx]);
            v = fmaxf(v * scs[ci] + shs[ci], 0.f);
            hv = f2bf(v);
        }
        x2[(row * 64 + ci) * 98 + x] = hv;
    }
    __syncthreads();
    if (tid < 96) {
        float acc = c2b[0];
        for (int ci = 0; ci < 64; ci++)
            for (int ky = 0; ky < 3; ky++)
                for (int kx = 0; kx < 3; kx++)
                    acc += bf2f(x2[(ky * 64 + ci) * 98 + tid + kx]) * w2s[ci * 9 + ky * 3 + kx];
        mskout[(size_t)bn * HWp + y * 96 + tid] = acc;
    }
}

// =========== spatial softmax + center-of-mass loss ===========
__global__ void k_smloss(const float* __restrict__ msk, float* loss) {
    int bn = blockIdx.x, tid = threadIdx.x;
    __shared__ float cof[HWp];
    __shared__ float red[256];
    const float* m = msk + (size_t)bn * HWp;
    const float st = 2.f / 95.f;
    float lmax = -1e30f;
    for (int p = tid; p < HWp; p += 256) { float v = m[p]; cof[p] = v; lmax = fmaxf(lmax, v); }
    red[tid] = lmax; __syncthreads();
    for (int s = 128; s > 0; s >>= 1) { if (tid < s) red[tid] = fmaxf(red[tid], red[tid+s]); __syncthreads(); }
    float mx = red[0]; __syncthreads();
    float lsum = 0.f;
    for (int p = tid; p < HWp; p += 256) { float e = __expf(cof[p] - mx); cof[p] = e; lsum += e; }
    red[tid] = lsum; __syncthreads();
    for (int s = 128; s > 0; s >>= 1) { if (tid < s) red[tid] += red[tid+s]; __syncthreads(); }
    float inv = 1.f / red[0]; __syncthreads();
    float cy = 0.f, cx = 0.f;
    for (int p = tid; p < HWp; p += 256) {
        float c = cof[p] * inv; cof[p] = c;
        float iy = -1.f + st * (float)(p / 96), ix = -1.f + st * (float)(p % 96);
        cy += c * iy; cx += c * ix;
    }
    red[tid] = cy; __syncthreads();
    for (int s = 128; s > 0; s >>= 1) { if (tid < s) red[tid] += red[tid+s]; __syncthreads(); }
    cy = red[0]; __syncthreads();
    red[tid] = cx; __syncthreads();
    for (int s = 128; s > 0; s >>= 1) { if (tid < s) red[tid] += red[tid+s]; __syncthreads(); }
    cx = red[0]; __syncthreads();
    float lp = 0.f;
    for (int p = tid; p < HWp; p += 256) {
        float iy = -1.f + st * (float)(p / 96), ix = -1.f + st * (float)(p % 96);
        float dy = iy - cy, dx = ix - cx;
        lp += cof[p] * (dy * dy + dx * dx);
    }
    red[tid] = lp; __syncthreads();
    for (int s = 128; s > 0; s >>= 1) { if (tid < s) red[tid] += red[tid+s]; __syncthreads(); }
    if (tid == 0) atomicAdd(loss, red[0] * (1.f / 128.f));
}

// =========== instance-norm 1 over msk ===========
__global__ void k_in1(const float* __restrict__ msk, const float* g, const float* b_,
                      float* sc, float* sh) {
    int bn = blockIdx.x, n = bn & 15, tid = threadIdx.x;
    const float* m = msk + (size_t)bn * HWp;
    float s = 0.f, ss = 0.f;
    for (int p = tid; p < HWp; p += 256) { float v = m[p]; s += v; ss += v * v; }
    __shared__ float r0[256], r1[256];
    r0[tid] = s; r1[tid] = ss; __syncthreads();
    for (int st = 128; st > 0; st >>= 1) {
        if (tid < st) { r0[tid] += r0[tid+st]; r1[tid] += r1[tid+st]; }
        __syncthreads();
    }
    if (tid == 0) {
        float mu = r0[0] / (float)HWp, var = r1[0] / (float)HWp - mu * mu;
        float scv = g[n] * rsqrtf(var + EPSV);
        sc[bn] = scv; sh[bn] = b_[n] - mu * scv;
    }
}

// =========== gen-block 1x1 conv 16->16 + IN2 stats ===========
__global__ void k_g2(const float* __restrict__ msk, const float* sc, const float* sh,
                     const float* c3w, const float* c3b, float* g2out, float* st2) {
    int b = blockIdx.x / 36;
    int p = (blockIdx.x % 36) * 256 + threadIdx.x;
    int tid = threadIdx.x, lane = tid & 31;
    __shared__ float w[256], bb[16], scl[16], shl[16];
    w[tid] = c3w[tid];
    if (tid < 16) { bb[tid] = c3b[tid]; scl[tid] = sc[b * 16 + tid]; shl[tid] = sh[b * 16 + tid]; }
    __syncthreads();
    float act[16];
    for (int n = 0; n < 16; n++)
        act[n] = fmaxf(msk[((size_t)(b * 16 + n)) * HWp + p] * scl[n] + shl[n], 0.f);
    for (int o = 0; o < 16; o++) {
        float a = bb[o];
        for (int n = 0; n < 16; n++) a += w[o * 16 + n] * act[n];
        g2out[((size_t)(b * 16 + o)) * HWp + p] = a;
        float s = a, ss = a * a;
        for (int off = 16; off > 0; off >>= 1) {
            s += __shfl_down(s, off, 32); ss += __shfl_down(ss, off, 32);
        }
        if (lane == 0) {
            atomicAdd(&st2[(b * 16 + o) * 2 + 0], s);
            atomicAdd(&st2[(b * 16 + o) * 2 + 1], ss);
        }
    }
}

__global__ void k_in2fin(const float* st2, const float* g, const float* b_,
                         float* sc, float* sh) {
    int i = threadIdx.x;  // 0..127
    int o = i & 15;
    float s = st2[i * 2], ss = st2[i * 2 + 1];
    float mu = s / (float)HWp, var = ss / (float)HWp - mu * mu;
    float scv = g[o] * rsqrtf(var + EPSV);
    sc[i] = scv; sh[i] = b_[o] - mu * scv;
}

// =========== gen-block 1x1 conv 16->3 + sigmoid ===========
__global__ void k_rec(const float* __restrict__ g2in, const float* sc, const float* sh,
                      const float* c4w, const float* c4b, float* rec) {
    int b = blockIdx.x / 36;
    int p = (blockIdx.x % 36) * 256 + threadIdx.x;
    int tid = threadIdx.x;
    __shared__ float w4[48], b4[3], scl[16], shl[16];
    if (tid < 48) w4[tid] = c4w[tid];
    if (tid < 3) b4[tid] = c4b[tid];
    if (tid < 16) { scl[tid] = sc[b * 16 + tid]; shl[tid] = sh[b * 16 + tid]; }
    __syncthreads();
    float act[16];
    for (int o = 0; o < 16; o++)
        act[o] = fmaxf(g2in[((size_t)(b * 16 + o)) * HWp + p] * scl[o] + shl[o], 0.f);
    for (int o3 = 0; o3 < 3; o3++) {
        float a = b4[o3];
        for (int o = 0; o < 16; o++) a += w4[o3 * 16 + o] * act[o];
        rec[((size_t)(b * 3 + o3)) * HWp + p] = 1.f / (1.f + __expf(-a));
    }
}

// =========== op_out = op[b,n,c] * msk[b,n,p] ===========
__global__ void k_opout(const float* __restrict__ op, const float* __restrict__ msk,
                        float* __restrict__ out) {
    size_t gid = (size_t)blockIdx.x * 256 + threadIdx.x;    // 18,874,368 quads
    int pq = (int)(gid % 2304);
    size_t r = gid / 2304;
    int c = (int)(r % 64);
    int bn = (int)(r / 64);
    float o = op[bn * 64 + c];
    float4 m4 = *(const float4*)(msk + (size_t)bn * HWp + pq * 4);
    float4 res = { o * m4.x, o * m4.y, o * m4.z, o * m4.w };
    *(float4*)(out + ((size_t)bn * 64 + c) * HWp + pq * 4) = res;
}

extern "C" void kernel_launch(void* const* d_in, const int* in_sizes, int n_in,
                              void* d_out, int out_size, void* d_ws, size_t ws_size,
                              hipStream_t stream) {
    const float* sr    = (const float*)d_in[0];
    const float* tg    = (const float*)d_in[1];
    const float* w_w   = (const float*)d_in[2];
    const float* w_b   = (const float*)d_in[3];
    const float* gn1_g = (const float*)d_in[4];
    const float* gn1_b = (const float*)d_in[5];
    const float* c1_w  = (const float*)d_in[6];
    const float* c1_b  = (const float*)d_in[7];
    const float* gn2_g = (const float*)d_in[8];
    const float* gn2_b = (const float*)d_in[9];
    const float* c2_w  = (const float*)d_in[10];
    const float* c2_b  = (const float*)d_in[11];
    const float* in1_g = (const float*)d_in[12];
    const float* in1_b = (const float*)d_in[13];
    const float* c3_w  = (const float*)d_in[14];
    const float* c3_b  = (const float*)d_in[15];
    const float* in2_g = (const float*)d_in[16];
    const float* in2_b = (const float*)d_in[17];
    const float* c4_w  = (const float*)d_in[18];
    const float* c4_b  = (const float*)d_in[19];

    float* out = (float*)d_out;
    float* wsf = (float*)d_ws;
    float* tg_sum = wsf + WS_TGSUM;
    float* tg_ss  = wsf + WS_TGSS;
    float* sr_sum = wsf + WS_SRSUM;
    float* op_ws  = wsf + WS_OP;
    float* sc1    = wsf + WS_SC1;
    float* sh1    = wsf + WS_SH1;
    float* gn2st  = wsf + WS_GN2ST;
    float* sc2    = wsf + WS_SC2;
    float* sh2    = wsf + WS_SH2;
    float* in1sc  = wsf + WS_IN1SC;
    float* in1sh  = wsf + WS_IN1SH;
    float* in2st  = wsf + WS_IN2ST;
    float* in2sc  = wsf + WS_IN2SC;
    float* in2sh  = wsf + WS_IN2SH;
    float* g2buf  = wsf + WS_G2;
    unsigned short* wA = (unsigned short*)(wsf + WS_WA);
    unsigned short* y1 = (unsigned short*)(wsf + WS_Y1);

    k_zero<<<17, 256, 0, stream>>>(gn2st, in2st, out + OUT_LOSS);
    k_stats<<<512, 256, 0, stream>>>(tg, sr, tg_sum, tg_ss, sr_sum);
    k_packw1<<<144, 256, 0, stream>>>(c1_w, wA);
    k_routing<<<BN, 256, 0, stream>>>(sr, w_w, w_b, sr_sum, op_ws, out + OUT_COF);
    k_gn1prep<<<BN, 64, 0, stream>>>(tg_sum, tg_ss, op_ws, gn1_g, gn1_b, sc1, sh1);
    k_conv1<<<BN * HH, 256, 0, stream>>>(tg, sc1, sh1, c1_b, wA, y1, gn2st);
    k_gn2fin<<<BN, 64, 0, stream>>>(gn2st, gn2_g, gn2_b, sc2, sh2);
    k_conv2<<<BN * HH, 256, 0, stream>>>(y1, sc2, sh2, c2_w, c2_b, out + OUT_MSK);
    k_smloss<<<BN, 256, 0, stream>>>(out + OUT_MSK, out + OUT_LOSS);
    k_in1<<<BN, 256, 0, stream>>>(out + OUT_MSK, in1_g, in1_b, in1sc, in1sh);
    k_g2<<<Bb * 36, 256, 0, stream>>>(out + OUT_MSK, in1sc, in1sh, c3_w, c3_b, g2buf, in2st);
    k_in2fin<<<1, 128, 0, stream>>>(in2st, in2_g, in2_b, in2sc, in2sh);
    k_rec<<<Bb * 36, 256, 0, stream>>>(g2buf, in2sc, in2sh, c4_w, c4_b, out + OUT_REC);
    k_opout<<<73728, 256, 0, stream>>>(op_ws, out + OUT_MSK, out + OUT_OPOUT);
}
